// BasicTransformer_18940805775699
// MI455X (gfx1250) — compile-verified, matrix operand order checked by reading, not run on hardware
//
#include <hip/hip_runtime.h>
#include <hip/hip_bf16.h>
#include <math.h>

// ---------------------------------------------------------------------------
// MI455X (gfx1250) transformer block: pre-LN attn + pre-LN MLP, f16 WMMA path
// ---------------------------------------------------------------------------
typedef __attribute__((ext_vector_type(16))) _Float16 v16h;
typedef __attribute__((ext_vector_type(8)))  _Float16 v8h;
typedef __attribute__((ext_vector_type(8)))  float    v8f;
typedef __attribute__((ext_vector_type(4)))  int      v4i;

constexpr int kB   = 2;
constexpr int kS   = 2048;
constexpr int kH   = 2048;
constexpr int kNH  = 32;
constexpr int kHD  = 64;
constexpr int kFFN = 8192;
constexpr int kBS  = kB * kS;          // 4096 rows
constexpr int kQKV = 3 * kH;           // 6144

// ---- CDNA5 feature probes (compile-safe: fall back if builtin is absent) ---
#if defined(__HIP_DEVICE_COMPILE__) && __has_builtin(__builtin_amdgcn_global_load_async_to_lds_b128)
#define HAS_ASYNC_LDS 1
#else
#define HAS_ASYNC_LDS 0
#endif
#if defined(__HIP_DEVICE_COMPILE__) && __has_builtin(__builtin_amdgcn_permlane16)
#define HAS_PERMLANE 1
#else
#define HAS_PERMLANE 0
#endif

#if HAS_ASYNC_LDS
// Pointer types matching the builtin's signature: v4i in addrspace(1)/(3).
typedef __attribute__((address_space(1))) v4i* gptr4;
typedef __attribute__((address_space(3))) v4i* lptr4;
// 16-byte async copy global -> LDS (per-lane addresses, tracked by ASYNCcnt)
__device__ inline void async_copy16(const _Float16* g, _Float16* l) {
  __builtin_amdgcn_global_load_async_to_lds_b128(
      (gptr4)(size_t)g, (lptr4)(unsigned int)(size_t)l, 0, 0);
}
__device__ inline void async_wait4() {
#if __has_builtin(__builtin_amdgcn_s_wait_asynccnt)
  __builtin_amdgcn_s_wait_asynccnt(4);
#else
  asm volatile("s_wait_asynccnt 0x4" ::: "memory");
#endif
}
__device__ inline void async_wait0() {
#if __has_builtin(__builtin_amdgcn_s_wait_asynccnt)
  __builtin_amdgcn_s_wait_asynccnt(0);
#else
  asm volatile("s_wait_asynccnt 0x0" ::: "memory");
#endif
}
#endif

// ---- xor-shuffle reductions within each 16-lane half (rows of C/D layout) --
#if HAS_PERMLANE
__device__ inline float pl16(float v, unsigned lo, unsigned hi) {
  unsigned u = __builtin_bit_cast(unsigned, v);
  u = __builtin_amdgcn_permlane16(u, u, lo, hi, false, false);
  return __builtin_bit_cast(float, u);
}
#endif
__device__ inline float red_max16(float v) {
#if HAS_PERMLANE
  v = fmaxf(v, pl16(v, 0x67452301u, 0xEFCDAB89u));   // i^1
  v = fmaxf(v, pl16(v, 0x54761032u, 0xDCFE98BAu));   // i^2
  v = fmaxf(v, pl16(v, 0x32107654u, 0xBA98FEDCu));   // i^4
  v = fmaxf(v, pl16(v, 0xFEDCBA98u, 0x76543210u));   // i^8
#else
#pragma unroll
  for (int off = 1; off < 16; off <<= 1) v = fmaxf(v, __shfl_xor(v, off, 32));
#endif
  return v;
}
__device__ inline float red_sum16(float v) {
#if HAS_PERMLANE
  v += pl16(v, 0x67452301u, 0xEFCDAB89u);
  v += pl16(v, 0x54761032u, 0xDCFE98BAu);
  v += pl16(v, 0x32107654u, 0xBA98FEDCu);
  v += pl16(v, 0xFEDCBA98u, 0x76543210u);
#else
#pragma unroll
  for (int off = 1; off < 16; off <<= 1) v += __shfl_xor(v, off, 32);
#endif
  return v;
}

// Build a 16-bit WMMA A/B fragment (16x32, row striped in lane) from a row
// pointer whose K=0..31 block starts at p.  Elements 0..7 = K = kh..kh+7,
// elements 8..15 = K = 16+kh..16+kh+7  (kh = (lane>>4)*8).
__device__ inline v16h mk_frag(const _Float16* p, int kh) {
  union { v16h v; v8h h[2]; } u;
  u.h[0] = *(const v8h*)(p + kh);
  u.h[1] = *(const v8h*)(p + kh + 16);
  return u.v;
}

__device__ inline float gelu_tanh(float x) {
  return 0.5f * x * (1.0f + tanhf(0.7978845608f * (x + 0.044715f * x * x * x)));
}

// ---------------------------------------------------------------------------
// fp32 [K,N] -> f16 [N,K] transposed weight conversion (32x32 LDS tiles)
// ---------------------------------------------------------------------------
__global__ __launch_bounds__(256) void transpose_f16_kernel(
    const float* __restrict__ in, _Float16* __restrict__ out, int K, int N) {
  __shared__ _Float16 t[32][33];
  const int tx = threadIdx.x & 31;
  const int ty = threadIdx.x >> 5;               // 0..7
  const int k0 = blockIdx.y * 32, n0 = blockIdx.x * 32;
#pragma unroll
  for (int i = 0; i < 4; ++i)
    t[ty + i * 8][tx] = (_Float16)in[(size_t)(k0 + ty + i * 8) * N + n0 + tx];
  __syncthreads();
#pragma unroll
  for (int i = 0; i < 4; ++i)
    out[(size_t)(n0 + ty + i * 8) * K + k0 + tx] = t[tx][ty + i * 8];
}

// ---------------------------------------------------------------------------
// Row LayerNorm (H=2048) -> f16 output, one 256-thread block per row
// ---------------------------------------------------------------------------
__global__ __launch_bounds__(256) void ln_f16_kernel(
    const float* __restrict__ x, const float* __restrict__ g,
    const float* __restrict__ beta, _Float16* __restrict__ out) {
  const int row = blockIdx.x;
  const float* xr = x + (size_t)row * kH;
  float sum = 0.f, sq = 0.f;
  for (int i = threadIdx.x; i < kH; i += 256) {
    float v = xr[i];
    sum += v; sq += v * v;
  }
#pragma unroll
  for (int off = 16; off; off >>= 1) {
    sum += __shfl_xor(sum, off, 32);
    sq  += __shfl_xor(sq,  off, 32);
  }
  __shared__ float s0[8], s1[8];
  const int w = threadIdx.x >> 5, lane = threadIdx.x & 31;
  if (lane == 0) { s0[w] = sum; s1[w] = sq; }
  __syncthreads();
  if (threadIdx.x == 0) {
    float a = 0.f, b = 0.f;
    for (int i = 0; i < 8; ++i) { a += s0[i]; b += s1[i]; }
    s0[0] = a; s1[0] = b;
  }
  __syncthreads();
  const float mean = s0[0] * (1.0f / kH);
  const float var  = s1[0] * (1.0f / kH) - mean * mean;
  const float inv  = rsqrtf(var + 1e-5f);
  _Float16* orow = out + (size_t)row * kH;
  for (int i = threadIdx.x; i < kH; i += 256)
    orow[i] = (_Float16)((xr[i] - mean) * inv * g[i] + beta[i]);
}

// ---------------------------------------------------------------------------
// GEMM: C[M,N] = epilogue(A_f16[M,K] @ Wt_f16[N,K]^T + bias [+ res])
// 128x128x32 block tile, 8 waves, wave tile 32x64 (2x4 WMMA accumulators).
// Async double-buffered LDS staging when global_load_async_to_lds exists.
// ---------------------------------------------------------------------------
template <bool OUT_F16, bool GELU_ACT, bool HAS_RES>
__global__ __launch_bounds__(256) void wmma_gemm_kernel(
    const _Float16* __restrict__ A, const _Float16* __restrict__ Wt,
    const float* __restrict__ bias, const float* __restrict__ res,
    float* __restrict__ outF, _Float16* __restrict__ outH,
    int M, int N, int K) {
  constexpr int BM = 128, BN = 128, BK = 32, BKP = 40;  // 20-bank row stride
#if HAS_ASYNC_LDS
  constexpr int NBUF = 2;
#else
  constexpr int NBUF = 1;
#endif
  __shared__ _Float16 As[NBUF][BM][BKP];   // [m][k]
  __shared__ _Float16 Bs[NBUF][BN][BKP];   // [n][k]
  const int tid  = threadIdx.x;
  const int lane = tid & 31;
  const int wave = tid >> 5;
  const int wm = (wave & 3) * 32;
  const int wn = (wave >> 2) * 64;
  const int mBase = blockIdx.y * BM;
  const int nBase = blockIdx.x * BN;

  v8f acc[2][4];
#pragma unroll
  for (int i = 0; i < 2; ++i)
#pragma unroll
    for (int j = 0; j < 4; ++j) { v8f z = {}; acc[i][j] = z; }

  const int arow = tid >> 1, acol = (tid & 1) * 16;   // two 16B chunks / thread
  const int kh = (lane >> 4) * 8;
  const int kTiles = K / BK;

#if HAS_ASYNC_LDS
  auto issue = [&](int buf, int kt) {
    const _Float16* ag = A  + (size_t)(mBase + arow) * K + kt * BK + acol;
    const _Float16* wg = Wt + (size_t)(nBase + arow) * K + kt * BK + acol;
    async_copy16(ag,     &As[buf][arow][acol]);
    async_copy16(ag + 8, &As[buf][arow][acol + 8]);
    async_copy16(wg,     &Bs[buf][arow][acol]);
    async_copy16(wg + 8, &Bs[buf][arow][acol + 8]);
  };
  issue(0, 0);
#endif

  for (int kt = 0; kt < kTiles; ++kt) {
#if HAS_ASYNC_LDS
    const int cur = kt & 1;
    if (kt + 1 < kTiles) { issue(cur ^ 1, kt + 1); async_wait4(); }
    else                 { async_wait0(); }
#else
    const int cur = 0;
    const _Float16* ag = A  + (size_t)(mBase + arow) * K + kt * BK + acol;
    const _Float16* wg = Wt + (size_t)(nBase + arow) * K + kt * BK + acol;
    *(v8h*)&As[0][arow][acol]     = *(const v8h*)ag;
    *(v8h*)&As[0][arow][acol + 8] = *(const v8h*)(ag + 8);
    *(v8h*)&Bs[0][arow][acol]     = *(const v8h*)wg;
    *(v8h*)&Bs[0][arow][acol + 8] = *(const v8h*)(wg + 8);
    if (kt + 1 < kTiles) {
      __builtin_prefetch(ag + BK, 0, 1);
      __builtin_prefetch(wg + BK, 0, 1);
    }
#endif
    __syncthreads();

    v16h af[2], bf[4];
#pragma unroll
    for (int i = 0; i < 2; ++i)
      af[i] = mk_frag(&As[cur][wm + i * 16 + (lane & 15)][0], kh);
#pragma unroll
    for (int j = 0; j < 4; ++j)
      bf[j] = mk_frag(&Bs[cur][wn + j * 16 + (lane & 15)][0], kh);
#pragma unroll
    for (int i = 0; i < 2; ++i)
#pragma unroll
      for (int j = 0; j < 4; ++j)
        acc[i][j] = __builtin_amdgcn_wmma_f32_16x16x32_f16(
            false, af[i], false, bf[j], (short)0, acc[i][j], false, false);
    __syncthreads();
  }

  // Epilogue: C/D layout -> col = lane&15, rows = r + (lane>>4)*8
  const int cn = lane & 15, rb = (lane >> 4) * 8;
#pragma unroll
  for (int i = 0; i < 2; ++i)
#pragma unroll
    for (int j = 0; j < 4; ++j) {
      const int n = nBase + wn + j * 16 + cn;
      const float bb = bias[n];
#pragma unroll
      for (int r = 0; r < 8; ++r) {
        const int m = mBase + wm + i * 16 + rb + r;
        float v = acc[i][j][r] + bb;
        if (GELU_ACT) v = gelu_tanh(v);
        if (HAS_RES)  v += res[(size_t)m * N + n];
        if (OUT_F16)  outH[(size_t)m * N + n] = (_Float16)v;
        else          outF[(size_t)m * N + n] = v;
      }
    }
}

// ---------------------------------------------------------------------------
// Flash attention: block = (b, head, 128 q rows), 8 waves x 16 q rows each.
// K/V 32-key tiles staged in LDS; online softmax; P relaid D->A via LDS.
// ---------------------------------------------------------------------------
__global__ __launch_bounds__(256) void flash_attn_kernel(
    const _Float16* __restrict__ qkv, _Float16* __restrict__ ctxOut) {
  __shared__ _Float16 Kt[32][72];       // [key][d]
  __shared__ _Float16 Vt[64][40];       // [d][key]  transposed for B-frags
  __shared__ _Float16 Ps[8][16][32];    // per-wave P scratch
  const int tid  = threadIdx.x;
  const int lane = tid & 31;
  const int wave = tid >> 5;
  const int bh = blockIdx.y;
  const int b = bh / kNH, h = bh % kNH;
  const int qrow0 = blockIdx.x * 128 + wave * 16;
  const int kh = (lane >> 4) * 8;
  const int rb = (lane >> 4) * 8;
  const int cn = lane & 15;

  const size_t qbase = ((size_t)(b * kS + qrow0 + cn)) * kQKV + (size_t)h * (3 * kHD);
  v16h qa[2];
#pragma unroll
  for (int c = 0; c < 2; ++c) qa[c] = mk_frag(qkv + qbase + c * 32, kh);

  v8f ctx[4];
#pragma unroll
  for (int j = 0; j < 4; ++j) { v8f z = {}; ctx[j] = z; }
  float mrow[8], lrow[8];
#pragma unroll
  for (int r = 0; r < 8; ++r) { mrow[r] = -3.0e38f; lrow[r] = 0.f; }

  const int key = tid >> 3;            // 0..31
  const int d0  = (tid & 7) * 8;       // 0..56

  for (int kt = 0; kt < kS / 32; ++kt) {
    const size_t rowOff = ((size_t)(b * kS + kt * 32 + key)) * kQKV + (size_t)h * 192;
    *(v8h*)&Kt[key][d0] = *(const v8h*)(qkv + rowOff + 64 + d0);
    v8h vv = *(const v8h*)(qkv + rowOff + 128 + d0);
#pragma unroll
    for (int i = 0; i < 8; ++i) Vt[d0 + i][key] = vv[i];
    __syncthreads();

    v8f sc[2];
#pragma unroll
    for (int g = 0; g < 2; ++g) {
      const _Float16* krow = &Kt[g * 16 + cn][0];
      v8f z = {};
      z = __builtin_amdgcn_wmma_f32_16x16x32_f16(false, qa[0], false, mk_frag(krow, kh),      (short)0, z, false, false);
      z = __builtin_amdgcn_wmma_f32_16x16x32_f16(false, qa[1], false, mk_frag(krow + 32, kh), (short)0, z, false, false);
      sc[g] = z;
    }

    float tmax[8], psum[8];
#pragma unroll
    for (int r = 0; r < 8; ++r) {
      sc[0][r] *= 0.125f; sc[1][r] *= 0.125f;
      tmax[r] = red_max16(fmaxf(sc[0][r], sc[1][r]));
    }
#pragma unroll
    for (int r = 0; r < 8; ++r) {
      const float mn = fmaxf(mrow[r], tmax[r]);
      const float alpha = __expf(mrow[r] - mn);
      mrow[r] = mn;
      lrow[r] *= alpha;
#pragma unroll
      for (int j = 0; j < 4; ++j) ctx[j][r] *= alpha;
      const float p0 = __expf(sc[0][r] - mn);
      const float p1 = __expf(sc[1][r] - mn);
      psum[r] = p0 + p1;
      Ps[wave][rb + r][cn]      = (_Float16)p0;
      Ps[wave][rb + r][16 + cn] = (_Float16)p1;
    }
#pragma unroll
    for (int r = 0; r < 8; ++r) lrow[r] += red_sum16(psum[r]);
    __syncthreads();   // Ps visible; all waves done with Kt

    const v16h pa = mk_frag(&Ps[wave][cn][0], kh);
#pragma unroll
    for (int j = 0; j < 4; ++j) {
      const v16h vb = mk_frag(&Vt[j * 16 + cn][0], kh);
      ctx[j] = __builtin_amdgcn_wmma_f32_16x16x32_f16(
          false, pa, false, vb, (short)0, ctx[j], false, false);
    }
    __syncthreads();   // protect Kt/Vt/Ps before next tile overwrites
  }

#pragma unroll
  for (int r = 0; r < 8; ++r) {
    const float inv = 1.0f / lrow[r];
    const int sq = qrow0 + rb + r;
    const size_t obase = ((size_t)(b * kS + sq)) * kH + (size_t)h * kHD;
#pragma unroll
    for (int j = 0; j < 4; ++j)
      ctxOut[obase + j * 16 + cn] = (_Float16)(ctx[j][r] * inv);
  }
}

// ---------------------------------------------------------------------------
extern "C" void kernel_launch(void* const* d_in, const int* in_sizes, int n_in,
                              void* d_out, int out_size, void* d_ws, size_t ws_size,
                              hipStream_t stream) {
  (void)in_sizes; (void)n_in; (void)out_size; (void)ws_size;
  const float* x     = (const float*)d_in[0];
  const float* ln1g  = (const float*)d_in[1];
  const float* ln1b  = (const float*)d_in[2];
  const float* wqkv  = (const float*)d_in[3];
  const float* bqkv  = (const float*)d_in[4];
  const float* wproj = (const float*)d_in[5];
  const float* bproj = (const float*)d_in[6];
  const float* ln2g  = (const float*)d_in[7];
  const float* ln2b  = (const float*)d_in[8];
  const float* w1    = (const float*)d_in[9];
  const float* b1    = (const float*)d_in[10];
  const float* w2    = (const float*)d_in[11];
  const float* b2    = (const float*)d_in[12];
  float* out = (float*)d_out;

  char* ws = (char*)d_ws;
  size_t off = 0;
  auto alloc = [&](size_t bytes) -> void* {
    void* p = ws + off;
    off += (bytes + 255) & ~(size_t)255;
    return p;
  };
  // f16 transposed weights [N][K]
  _Float16* wqkvT  = (_Float16*)alloc((size_t)kQKV * kH   * 2);
  _Float16* wprojT = (_Float16*)alloc((size_t)kH   * kH   * 2);
  _Float16* w1T    = (_Float16*)alloc((size_t)kFFN * kH   * 2);
  _Float16* w2T    = (_Float16*)alloc((size_t)kH   * kFFN * 2);
  // f16 activations / fp32 residual
  _Float16* ln1h = (_Float16*)alloc((size_t)kBS * kH   * 2);
  _Float16* qkvh = (_Float16*)alloc((size_t)kBS * kQKV * 2);
  _Float16* ctxh = (_Float16*)alloc((size_t)kBS * kH   * 2);
  float*    att  = (float*)   alloc((size_t)kBS * kH   * 4);
  _Float16* ln2h = (_Float16*)alloc((size_t)kBS * kH   * 2);
  _Float16* h1h  = (_Float16*)alloc((size_t)kBS * kFFN * 2);

  // 0) weights: fp32 [K,N] -> f16 [N,K]
  transpose_f16_kernel<<<dim3(kQKV / 32, kH / 32),   256, 0, stream>>>(wqkv,  wqkvT,  kH,   kQKV);
  transpose_f16_kernel<<<dim3(kH / 32,   kH / 32),   256, 0, stream>>>(wproj, wprojT, kH,   kH);
  transpose_f16_kernel<<<dim3(kFFN / 32, kH / 32),   256, 0, stream>>>(w1,    w1T,    kH,   kFFN);
  transpose_f16_kernel<<<dim3(kH / 32,   kFFN / 32), 256, 0, stream>>>(w2,    w2T,    kFFN, kH);

  // 1) LN1
  ln_f16_kernel<<<kBS, 256, 0, stream>>>(x, ln1g, ln1b, ln1h);
  // 2) QKV = ln1 @ w_qkv + b_qkv   -> f16
  wmma_gemm_kernel<true, false, false><<<dim3(kQKV / 128, kBS / 128), 256, 0, stream>>>(
      ln1h, wqkvT, bqkv, nullptr, nullptr, qkvh, kBS, kQKV, kH);
  // 3) flash attention -> ctx f16
  flash_attn_kernel<<<dim3(kS / 128, kB * kNH), 256, 0, stream>>>(qkvh, ctxh);
  // 4) attn_out = x + ctx @ w_proj + b_proj   -> f32
  wmma_gemm_kernel<false, false, true><<<dim3(kH / 128, kBS / 128), 256, 0, stream>>>(
      ctxh, wprojT, bproj, x, att, nullptr, kBS, kH, kH);
  // 5) LN2
  ln_f16_kernel<<<kBS, 256, 0, stream>>>(att, ln2g, ln2b, ln2h);
  // 6) h1 = gelu(ln2 @ w1 + b1)   -> f16
  wmma_gemm_kernel<true, true, false><<<dim3(kFFN / 128, kBS / 128), 256, 0, stream>>>(
      ln2h, w1T, b1, nullptr, nullptr, h1h, kBS, kFFN, kH);
  // 7) out = attn_out + h1 @ w2 + b2   -> f32
  wmma_gemm_kernel<false, false, true><<<dim3(kH / 128, kBS / 128), 256, 0, stream>>>(
      h1h, w2T, b2, att, out, nullptr, kBS, kH, kFFN);
}
